// MultiHeadSelfAttention_20220706030494
// MI455X (gfx1250) — matrix operand order repaired and checked by reading, not compiled, not run on hardware
//
#include <hip/hip_runtime.h>
#include <stdint.h>

// ---------------------------------------------------------------------------
// MHSA for MI455X (gfx1250, wave32, WMMA).
// B=4 S=2048 H=16 D=64 E=1024.  ~137 GFLOP -> compute-bound on bf16 WMMA.
// LDS-tiled double-buffered GEMM for both projections (87 FLOP/B of global
// traffic, barrier-pipelined), dual-q-tile flash attention (32 FLOP/B),
// balanced causal pairing, coalesced b128 epilogues.
// This round: two-source v_cvt_pk_bf16_f32 via __builtin_convertvector, and
// softmax scale (1/sqrt(E) * log2e) folded into Q so the exp path is just
// pk_add + v_exp (hardware exp2) with no per-element multiply.
// Workspace layout (88 MiB): x_bf16 | wq/wk/wv/wo_bf16 | Q | K | Vt | O
// ---------------------------------------------------------------------------

#define BQ 4
#define SQ 2048
#define HQ 16
#define DQ 64
#define EQ 1024
#define MROWS (BQ * SQ) /* 8192 */
#define APAD 40         /* LDS tile row stride (32 bf16 + 8 pad) */
#define QSCALE 0.04508422f /* (1/sqrt(1024)) * log2(e) */

typedef __attribute__((ext_vector_type(16))) __bf16 v16bf;
typedef __attribute__((ext_vector_type(2)))  __bf16 v2bf;
typedef __attribute__((ext_vector_type(2)))  float  v2f;
typedef __attribute__((ext_vector_type(8)))  float  v8f;

#if __has_builtin(__builtin_amdgcn_exp2f)
#define EXP2(x) __builtin_amdgcn_exp2f(x)
#else
#define EXP2(x) exp2f(x)
#endif

__device__ __forceinline__ unsigned short f2bf(float f) {
  union { __bf16 b; unsigned short u; } x;
  x.b = (__bf16)f;                       // RNE fptrunc -> native bf16 cvt
  return x.u;
}

__device__ __forceinline__ unsigned pack2bf(float lo, float hi) {
  v2f f; f.x = lo; f.y = hi;
  union { v2bf b; unsigned u; } x;
  x.b = __builtin_convertvector(f, v2bf); // single v_cvt_pk_bf16_f32
  return x.u;
}

__device__ __forceinline__ v8f vzero8() {
  v8f z;
#pragma unroll
  for (int i = 0; i < 8; ++i) z[i] = 0.0f;
  return z;
}

union FragAB {               // 16x32 (A, MxK) or 32x16 (B, KxN) bf16 fragment
  v16bf v;
  uint4 q[2];
  unsigned u[8];
  unsigned short h[16];
};

// A-matrix 16x32 bf16: lane row M=lane&15; 16 contiguous K at kbase+kh*16.
__device__ __forceinline__ void loadA(FragAB& f, const unsigned short* base,
                                      int row, int ld, int kbase, int kh) {
  const unsigned short* p = base + (size_t)row * ld + kbase + kh * 16;
  f.q[0] = *(const uint4*)(p);
  f.q[1] = *(const uint4*)(p + 8);
}

// B-matrix 32x16 bf16: lane col N=lane&15; 16 contiguous K at kbase+kh*16.
__device__ __forceinline__ void loadB(FragAB& f, const unsigned short* base,
                                      int row, int ld, int kbase, int kh) {
  const unsigned short* p = base + (size_t)row * ld + kbase + kh * 16;
  f.q[0] = *(const uint4*)(p);
  f.q[1] = *(const uint4*)(p + 8);
}

__device__ __forceinline__ v8f wmma_bf(const FragAB& a, const FragAB& b, v8f c) {
  return __builtin_amdgcn_wmma_f32_16x16x32_bf16(
      false, a.v, false, b.v, (short)0, c, false, false);
}

// ---------------------------------------------------------------------------
// fp32 -> bf16 conversion, 8 elements/thread (b128 in, b128 out)
// ---------------------------------------------------------------------------
__global__ void cvt_bf16_v8(const float* __restrict__ in,
                            unsigned short* __restrict__ out, int n8) {
  int i = blockIdx.x * blockDim.x + threadIdx.x;
  int stride = gridDim.x * blockDim.x;
  for (; i < n8; i += stride) {
    const float4* p = (const float4*)in + 2 * (size_t)i;
    float4 f0 = p[0], f1 = p[1];
    uint4 o;
    o.x = pack2bf(f0.x, f0.y);
    o.y = pack2bf(f0.z, f0.w);
    o.z = pack2bf(f1.x, f1.y);
    o.w = pack2bf(f1.z, f1.w);
    ((uint4*)out)[i] = o;
  }
}

// ---------------------------------------------------------------------------
// LDS-tiled GEMM  C = A(MxK) @ Bw(NxK)^T, bf16 in, f32 accum.
// WG = 8 waves (2m x 4n), WG tile 128x256, wave tile 64x64, K step 32,
// double-buffered LDS (one s_barrier per k-step).
// MODE 0: QKV (Bw = stacked wq|wk|wv, N=3072) -> Q/K [b][h][s][d], Vt
//         [b][h][d][s] via LDS-transposed b128 epilogue. Q pre-scaled by
//         QSCALE so attention needs no per-element scale multiply.
// MODE 1: out-proj (N=1024) -> f32 nontemporal stores.
// ---------------------------------------------------------------------------
template <int MODE>
__global__ __launch_bounds__(256) void gemm_tiled(
    const unsigned short* __restrict__ A,
    const unsigned short* __restrict__ Bw,
    unsigned short* __restrict__ Qb,
    unsigned short* __restrict__ Kb,
    unsigned short* __restrict__ Vt,
    float* __restrict__ outF,
    int NB /* N/256 */) {
  __shared__ __align__(16) unsigned short smem[30720]; // 60 KiB
  // scalar LDS pointers only (aggregate init of LDS-derived generic pointers
  // produces addrspacecast static initializers that ld.lld rejects)
  unsigned short* curA = smem;
  unsigned short* nxtA = smem + 5120;            // 128*APAD each
  unsigned short* curB = smem + 10240;
  unsigned short* nxtB = smem + 20480;           // 256*APAD each

  int tid = threadIdx.x, lane = tid & 31, wvi = tid >> 5;
  int mblk = blockIdx.x / NB, nblk = blockIdx.x % NB;
  int mwg = mblk * 128, nwg = nblk * 256;
  int wm = wvi >> 2, wn = wvi & 3;          // wave grid 2 x 4
  int col = lane & 15, kh = lane >> 4;

  // staging addresses for this thread
  int rowA = tid >> 1, halfA = tid & 1;     // A: 128 rows, 32B each half
  const unsigned short* gA = A + (size_t)(mwg + rowA) * EQ + halfA * 16;
  const unsigned short* gB = Bw + (size_t)(nwg + tid) * EQ; // B: 256 rows, 64B
  int sAoff = rowA * APAD + halfA * 16;
  int sBoff = tid * APAD;

  v8f acc[4][4];
#pragma unroll
  for (int i = 0; i < 4; ++i)
#pragma unroll
    for (int j = 0; j < 4; ++j) acc[i][j] = vzero8();

  // prologue: stage k-chunk 0
  {
    uint4 ra0 = *(const uint4*)(gA);
    uint4 ra1 = *(const uint4*)(gA + 8);
    uint4 rb0 = *(const uint4*)(gB);
    uint4 rb1 = *(const uint4*)(gB + 8);
    uint4 rb2 = *(const uint4*)(gB + 16);
    uint4 rb3 = *(const uint4*)(gB + 24);
    *(uint4*)(curA + sAoff) = ra0;
    *(uint4*)(curA + sAoff + 8) = ra1;
    *(uint4*)(curB + sBoff) = rb0;
    *(uint4*)(curB + sBoff + 8) = rb1;
    *(uint4*)(curB + sBoff + 16) = rb2;
    *(uint4*)(curB + sBoff + 24) = rb3;
  }
  __syncthreads();

  for (int kc = 0; kc < EQ; kc += 32) {
    uint4 ra0, ra1, rb0, rb1, rb2, rb3;
    bool more = (kc + 32 < EQ);
    if (more) {
      const unsigned short* ga = gA + kc + 32;
      const unsigned short* gb = gB + kc + 32;
      ra0 = *(const uint4*)(ga);      ra1 = *(const uint4*)(ga + 8);
      rb0 = *(const uint4*)(gb);      rb1 = *(const uint4*)(gb + 8);
      rb2 = *(const uint4*)(gb + 16); rb3 = *(const uint4*)(gb + 24);
    }
    // compute from current buffers
    FragAB af[4];
#pragma unroll
    for (int i = 0; i < 4; ++i)
      loadA(af[i], curA, wm * 64 + i * 16 + col, APAD, 0, kh);
#pragma unroll
    for (int j = 0; j < 4; ++j) {
      FragAB bfr;
      loadB(bfr, curB, wn * 64 + j * 16 + col, APAD, 0, kh);
#pragma unroll
      for (int i = 0; i < 4; ++i) acc[i][j] = wmma_bf(af[i], bfr, acc[i][j]);
    }
    if (more) {
      *(uint4*)(nxtA + sAoff) = ra0;      *(uint4*)(nxtA + sAoff + 8) = ra1;
      *(uint4*)(nxtB + sBoff) = rb0;      *(uint4*)(nxtB + sBoff + 8) = rb1;
      *(uint4*)(nxtB + sBoff + 16) = rb2; *(uint4*)(nxtB + sBoff + 24) = rb3;
    }
    __syncthreads();
    unsigned short* t;
    t = curA; curA = nxtA; nxtA = t;
    t = curB; curB = nxtB; nxtB = t;
  }

  // ------------------------- epilogue -------------------------
  int m0g = mwg + wm * 64;                  // 64 rows for this wave
  int n0g = nwg + wn * 64;                  // 64 cols for this wave

  if (MODE == 1) {
#pragma unroll
    for (int i = 0; i < 4; ++i)
#pragma unroll
      for (int j = 0; j < 4; ++j)
#pragma unroll
        for (int e = 0; e < 8; ++e)
          __builtin_nontemporal_store(
              acc[i][j][e],
              outF + (size_t)(m0g + i * 16 + kh * 8 + e) * EQ + n0g + j * 16 +
                  col);
    return;
  }

  // MODE 0: n0g in [0,3072): tensor = n0g>>10 (0=Q,1=K,2=V), head = bits 9:6
  int tensor = n0g >> 10;
  int head   = (n0g >> 6) & 15;
  int b      = m0g >> 11;
  int s0     = m0g & 2047;
  unsigned short* L = smem + wvi * 1536;    // wave-private staging region
  unsigned* L32 = (unsigned*)L;
  int srow = lane >> 1, half = lane & 1;

  if (tensor < 2) {
    // fold softmax scale * log2e into Q (scores then live in exp2 domain)
    float qs = (tensor == 0) ? QSCALE : 1.0f;
    unsigned short* gbase =
        (tensor == 0 ? Qb : Kb) + (size_t)(b * HQ + head) * SQ * DQ;
#pragma unroll
    for (int i = 0; i < 4; ++i) {           // per 16-row m-subtile
#pragma unroll
      for (int j = 0; j < 4; ++j)
#pragma unroll
        for (int e = 0; e < 8; ++e)
          L[(kh * 8 + e) * 72 + j * 16 + col] = f2bf(acc[i][j][e] * qs);
      // wave-private region; DS ops from one wave complete in order
      const uint4* src = (const uint4*)(L + srow * 72 + half * 32);
      uint4* dst =
          (uint4*)(gbase + (size_t)(s0 + i * 16 + srow) * DQ + half * 32);
      uint4 d0 = src[0], d1 = src[1], d2 = src[2], d3 = src[3];
      dst[0] = d0; dst[1] = d1; dst[2] = d2; dst[3] = d3;
    }
  } else {
    unsigned short* gbase = Vt + (size_t)(b * HQ + head) * DQ * SQ;
#pragma unroll
    for (int i = 0; i < 4; ++i) {           // per 16-col s-subtile
#pragma unroll
      for (int j = 0; j < 4; ++j) {
        int base = (j * 16 + col) * 24 + kh * 8; // even -> packed u32 stores
#pragma unroll
        for (int r = 0; r < 4; ++r)
          L32[(base >> 1) + r] =
              pack2bf(acc[i][j][2 * r], acc[i][j][2 * r + 1]);
      }
#pragma unroll
      for (int r = 0; r < 2; ++r) {
        int d = lane * 2 + r;
        const uint4* src = (const uint4*)(L + d * 24);
        uint4* dst = (uint4*)(gbase + (size_t)d * SQ + s0 + i * 16);
        uint4 d0 = src[0], d1 = src[1];
        dst[0] = d0; dst[1] = d1;
      }
    }
  }
}

// ---------------------------------------------------------------------------
// Flash attention, dual q-tile (32 q-rows) per wave so K/V fragments are
// reused twice (32 FLOP/B).  S^T = K @ Q^T keeps softmax state per-lane and
// exp'd probs become the P^T B-fragment with 8 packed shfl_xor(16) per q-tile.
// Q is pre-scaled by 1/sqrt(E)*log2e, so p = exp2(s - m): pk_add + v_exp only.
// O^T = Vt @ P^T, online softmax over 32-key steps.  Each wave processes the
// balanced causal block pair (qb, 63-qb).
// ---------------------------------------------------------------------------
__global__ __launch_bounds__(256) void attn_fwd(
    const unsigned short* __restrict__ Qb,
    const unsigned short* __restrict__ Kb,
    const unsigned short* __restrict__ Vt,
    unsigned short* __restrict__ Ob) {
  __shared__ __align__(16) unsigned short lds[8][16 * 72];

  int tid = threadIdx.x, lane = tid & 31, wvi = tid >> 5;
  int wid = blockIdx.x * 8 + wvi;    // 2048 waves, 2 blocks of 32 q-rows each
  int qbp = wid & 31;
  int h   = (wid >> 5) & 15;
  int b   = wid >> 9;
  int col = lane & 15, kh = lane >> 4;

  const unsigned short* Qh = Qb + (size_t)(b * HQ + h) * SQ * DQ;
  const unsigned short* Kh = Kb + (size_t)(b * HQ + h) * SQ * DQ;
  const unsigned short* Vh = Vt + (size_t)(b * HQ + h) * DQ * SQ;
  unsigned short* L = lds[wvi];
  unsigned* L32 = (unsigned*)L;

  for (int rep = 0; rep < 2; ++rep) {
    int qb = rep ? (63 - qbp) : qbp;
    int q0 = qb * 32;

    FragAB qf[2][2];                 // [u][dchunk], Q as B-matrix
    v8f o[2][4];
    float m[2], lsum[2];
#pragma unroll
    for (int u = 0; u < 2; ++u) {
      loadB(qf[u][0], Qh, q0 + u * 16 + col, DQ, 0, kh);
      loadB(qf[u][1], Qh, q0 + u * 16 + col, DQ, 32, kh);
      m[u] = -3.0e38f; lsum[u] = 0.0f;
#pragma unroll
      for (int t = 0; t < 4; ++t) o[u][t] = vzero8();
    }

    int nsteps = qb + 1;             // ((q0+31)>>5)+1
    for (int j = 0; j < nsteps; ++j) {
      int l0 = j * 32;
      // scores for both q-tiles, reusing K fragments
      FragAB kf0, kf1;
      v8f s[2][2];                   // [u][keytile]
      loadA(kf0, Kh, l0 + col, DQ, 0, kh);
      loadA(kf1, Kh, l0 + col, DQ, 32, kh);
#pragma unroll
      for (int u = 0; u < 2; ++u) {
        v8f z = vzero8();
        z = wmma_bf(kf0, qf[u][0], z);
        s[u][0] = wmma_bf(kf1, qf[u][1], z);
      }
      loadA(kf0, Kh, l0 + 16 + col, DQ, 0, kh);
      loadA(kf1, Kh, l0 + 16 + col, DQ, 32, kh);
#pragma unroll
      for (int u = 0; u < 2; ++u) {
        v8f z = vzero8();
        z = wmma_bf(kf0, qf[u][0], z);
        s[u][1] = wmma_bf(kf1, qf[u][1], z);
      }

      FragAB vf[4];
#pragma unroll
      for (int t = 0; t < 4; ++t) loadA(vf[t], Vh, t * 16 + col, SQ, l0, kh);
      if (j + 1 < nsteps) {
        __builtin_prefetch(Kh + (size_t)(l0 + 32 + col) * DQ + kh * 8, 0, 3);
        __builtin_prefetch(Vh + (size_t)col * SQ + l0 + 32, 0, 3);
      }

      FragAB pf[2];
#pragma unroll
      for (int u = 0; u < 2; ++u) {
        int q = q0 + u * 16 + col;
        float p0[8], p1[8], mloc = -3.0e38f;
#pragma unroll
        for (int i = 0; i < 8; ++i) {
          int key = l0 + kh * 8 + i; // S^T row = key, col = q
          float v0 = s[u][0][i]; if (key > q)      v0 = -3.0e38f;
          float v1 = s[u][1][i]; if (key + 16 > q) v1 = -3.0e38f;
          p0[i] = v0; p1[i] = v1;
          mloc = fmaxf(mloc, fmaxf(v0, v1));
        }
        mloc = fmaxf(mloc, __shfl_xor(mloc, 16, 32));
        float mn = fmaxf(m[u], mloc);
        float alpha = EXP2(m[u] - mn);   // log2 domain (QSCALE includes log2e)
        m[u] = mn;
        float ssum = 0.0f;
#pragma unroll
        for (int i = 0; i < 8; ++i) {
          p0[i] = EXP2(p0[i] - mn);
          p1[i] = EXP2(p1[i] - mn);
          ssum += p0[i] + p1[i];
        }
        ssum += __shfl_xor(ssum, 16, 32);
        lsum[u] = lsum[u] * alpha + ssum;
#pragma unroll
        for (int t = 0; t < 4; ++t)
#pragma unroll
          for (int i = 0; i < 8; ++i) o[u][t][i] *= alpha;
        // P^T (32keys x 16q) B-fragment: convert to bf16, pack pairs, then
        // swap row halves across the khalf pair with 8 packed shuffles.
        unsigned a0[4], a1[4];
#pragma unroll
        for (int r = 0; r < 4; ++r) {
          a0[r] = pack2bf(p0[2 * r], p0[2 * r + 1]);
          a1[r] = pack2bf(p1[2 * r], p1[2 * r + 1]);
        }
#pragma unroll
        for (int r = 0; r < 4; ++r) {
          unsigned b0 = __shfl_xor(a0[r], 16, 32);
          unsigned b1 = __shfl_xor(a1[r], 16, 32);
          pf[u].u[r]     = kh ? b1 : a0[r];
          pf[u].u[4 + r] = kh ? a1[r] : b0;
        }
      }
#pragma unroll
      for (int t = 0; t < 4; ++t) {
        o[0][t] = wmma_bf(vf[t], pf[0], o[0][t]);
        o[1][t] = wmma_bf(vf[t], pf[1], o[1][t]);
      }
    }

#pragma unroll
    for (int u = 0; u < 2; ++u) {
      float inv = 1.0f / lsum[u];
#pragma unroll
      for (int t = 0; t < 4; ++t) {
        int base = col * 72 + t * 16 + kh * 8;   // even -> packed u32 stores
#pragma unroll
        for (int r = 0; r < 4; ++r)
          L32[(base >> 1) + r] =
              pack2bf(o[u][t][2 * r] * inv, o[u][t][2 * r + 1] * inv);
      }
      // wave-private region; DS ops from one wave complete in order
      int qrow = lane >> 1, half = lane & 1;
      const uint4* src = (const uint4*)(L + qrow * 72 + half * 32);
      unsigned short* dst = Ob +
          (size_t)(b * SQ + q0 + u * 16 + qrow) * EQ + h * DQ + half * 32;
      uint4 d0 = src[0], d1 = src[1], d2 = src[2], d3 = src[3];
      ((uint4*)dst)[0] = d0; ((uint4*)dst)[1] = d1;
      ((uint4*)dst)[2] = d2; ((uint4*)dst)[3] = d3;
    }
  }
}

// ---------------------------------------------------------------------------
extern "C" void kernel_launch(void* const* d_in, const int* in_sizes, int n_in,
                              void* d_out, int out_size, void* d_ws,
                              size_t ws_size, hipStream_t stream) {
  const float* x  = (const float*)d_in[0];
  const float* wq = (const float*)d_in[1];
  const float* wk = (const float*)d_in[2];
  const float* wv = (const float*)d_in[3];
  const float* wo = (const float*)d_in[4];

  char* ws = (char*)d_ws;
  size_t off = 0;
  unsigned short* xb  = (unsigned short*)(ws + off); off += (size_t)MROWS * EQ * 2;
  unsigned short* wqb = (unsigned short*)(ws + off); off += (size_t)EQ * EQ * 2;
  unsigned short* wkb = (unsigned short*)(ws + off); off += (size_t)EQ * EQ * 2;
  unsigned short* wvb = (unsigned short*)(ws + off); off += (size_t)EQ * EQ * 2;
  unsigned short* wob = (unsigned short*)(ws + off); off += (size_t)EQ * EQ * 2;
  unsigned short* Qb  = (unsigned short*)(ws + off); off += (size_t)MROWS * EQ * 2;
  unsigned short* Kb  = (unsigned short*)(ws + off); off += (size_t)MROWS * EQ * 2;
  unsigned short* Vt  = (unsigned short*)(ws + off); off += (size_t)MROWS * EQ * 2;
  unsigned short* Ob  = (unsigned short*)(ws + off); off += (size_t)MROWS * EQ * 2;
  (void)ws_size; (void)in_sizes; (void)n_in; (void)out_size;

  cvt_bf16_v8<<<1024, 256, 0, stream>>>(x,  xb,  MROWS * EQ / 8);
  cvt_bf16_v8<<<256,  256, 0, stream>>>(wq, wqb, EQ * EQ / 8);
  cvt_bf16_v8<<<256,  256, 0, stream>>>(wk, wkb, EQ * EQ / 8);
  cvt_bf16_v8<<<256,  256, 0, stream>>>(wv, wvb, EQ * EQ / 8);
  cvt_bf16_v8<<<256,  256, 0, stream>>>(wo, wob, EQ * EQ / 8);

  // QKV: single GEMM vs stacked wq|wk|wv (contiguous in workspace), N=3072.
  gemm_tiled<0><<<64 * 12, 256, 0, stream>>>(xb, wqb, Qb, Kb, Vt, nullptr, 12);
  attn_fwd<<<256, 256, 0, stream>>>(Qb, Kb, Vt, Ob);
  // Out-proj: N=1024, f32 nontemporal stores to d_out.
  gemm_tiled<1><<<64 * 4, 256, 0, stream>>>(Ob, wob, nullptr, nullptr, nullptr,
                                            (float*)d_out, 4);
}
